// TransformerDecoder_3848290697591
// MI455X (gfx1250) — matrix operand order repaired and checked by reading, not compiled, not run on hardware
//
#include <hip/hip_runtime.h>
#include <hip/hip_bf16.h>

#define H 1024
#define NH 16
#define DH 64
#define FF 4096
#define LAYERS 4
#define BB 8
#define TT 512
#define SS 512
#define MROWS (BB * TT)   // 4096

typedef __attribute__((ext_vector_type(16))) __bf16 v16bf;
typedef __attribute__((ext_vector_type(2)))  __bf16 v2bf;
typedef __attribute__((ext_vector_type(8)))  float  v8f;

union BF16Frag { unsigned u[8]; v16bf v; };

__device__ __forceinline__ unsigned pack_bf16x2(float a, float b) {
    v2bf p;
    p.x = (__bf16)a;
    p.y = (__bf16)b;
    return __builtin_bit_cast(unsigned, p);   // lowers to v_cvt_pk_bf16_f32
}

// ---------------------------------------------------------------------------
// WMMA GEMM:  C[M,N] = act( A[M,K] @ W[N,K]^T + bias )
// A, W fp32 row-major (K contiguous). Block tile 128x128, K-slab 64, 8 waves.
// Each wave: 64x32 output = 4x2 fragments; 16 WMMAs per staged slab.
// Staging batches global loads first so the backend can clause them.
// ---------------------------------------------------------------------------
#define GTM 128
#define GTN 128
#define GTK 64
#define LDSW 36   // uints per LDS row: 32 (=64 bf16) + 4 pad (bank-conflict-free)

__global__ __launch_bounds__(256)
void k_gemm(const float* __restrict__ A, const float* __restrict__ W,
            const float* __restrict__ bias, float* __restrict__ C,
            int M, int N, int K, int relu)
{
    __shared__ unsigned As[GTM * LDSW];   // 18 KB
    __shared__ unsigned Ws[GTN * LDSW];   // 18 KB

    const int tid  = threadIdx.x;
    const int lane = tid & 31;
    const int wave = tid >> 5;
    const int wm   = wave >> 2;       // 0..1 -> M half (64 rows)
    const int wn   = wave & 3;        // 0..3 -> N quarter (32 cols)
    const int hf   = lane >> 4;       // half-wave
    const int l    = lane & 15;
    const int bm   = blockIdx.y * GTM;
    const int bn   = blockIdx.x * GTN;

    // Per-thread staging coordinates (fixed across K iterations)
    const int srow = tid >> 4;             // 0..15 base row
    const int skq  = (tid & 15) << 2;      // 0..60 k offset

    v8f acc[4][2];
    #pragma unroll
    for (int i = 0; i < 4; ++i)
        #pragma unroll
        for (int j = 0; j < 2; ++j)
            #pragma unroll
            for (int e = 0; e < 8; ++e) acc[i][j][e] = 0.0f;

    for (int k0 = 0; k0 < K; k0 += GTK) {
        // Phase 1: issue all global loads (A: 8 chunks, W: 8 chunks per thread)
        float4 fa[8], fw[8];
        #pragma unroll
        for (int it = 0; it < 8; ++it) {
            int r = srow + it * 16;
            fa[it] = *(const float4*)(A + (size_t)(bm + r) * K + k0 + skq);
        }
        #pragma unroll
        for (int it = 0; it < 8; ++it) {
            int r = srow + it * 16;
            const float* gp = W + (size_t)(bn + r) * K + k0 + skq;
            fw[it] = *(const float4*)gp;
            if (k0 + GTK < K) __builtin_prefetch(gp + GTK, 0, 1);  // global_prefetch_b8
        }
        // Phase 2: convert fp32 -> packed bf16 and store to LDS
        #pragma unroll
        for (int it = 0; it < 8; ++it) {
            int r = srow + it * 16;
            unsigned* d = &As[r * LDSW + (skq >> 1)];
            d[0] = pack_bf16x2(fa[it].x, fa[it].y);
            d[1] = pack_bf16x2(fa[it].z, fa[it].w);
        }
        #pragma unroll
        for (int it = 0; it < 8; ++it) {
            int r = srow + it * 16;
            unsigned* d = &Ws[r * LDSW + (skq >> 1)];
            d[0] = pack_bf16x2(fw[it].x, fw[it].y);
            d[1] = pack_bf16x2(fw[it].z, fw[it].w);
        }
        __syncthreads();

        #pragma unroll
        for (int ks = 0; ks < 2; ++ks) {      // two 16x16x32 K-steps per slab
            const int ko = ks * 16;            // uint offset into row
            // A fragment: lane (hf,l): M=l, K = {hf*8..+7, 16+hf*8..+7} of this step
            BF16Frag af[4], bf[2];
            #pragma unroll
            for (int mi = 0; mi < 4; ++mi) {
                int m = wm * 64 + mi * 16 + l;
                uint4 x0 = *(const uint4*)&As[m * LDSW + ko + hf * 4];
                uint4 x1 = *(const uint4*)&As[m * LDSW + ko + 8 + hf * 4];
                af[mi].u[0] = x0.x; af[mi].u[1] = x0.y; af[mi].u[2] = x0.z; af[mi].u[3] = x0.w;
                af[mi].u[4] = x1.x; af[mi].u[5] = x1.y; af[mi].u[6] = x1.z; af[mi].u[7] = x1.w;
            }
            // B fragment: lane (hf,l): N=l, K = {hf*8..+7, 16+hf*8..+7} of this step
            #pragma unroll
            for (int ni = 0; ni < 2; ++ni) {
                int n = wn * 32 + ni * 16 + l;
                uint4 x0 = *(const uint4*)&Ws[n * LDSW + ko + hf * 4];
                uint4 x1 = *(const uint4*)&Ws[n * LDSW + ko + 8 + hf * 4];
                bf[ni].u[0] = x0.x; bf[ni].u[1] = x0.y; bf[ni].u[2] = x0.z; bf[ni].u[3] = x0.w;
                bf[ni].u[4] = x1.x; bf[ni].u[5] = x1.y; bf[ni].u[6] = x1.z; bf[ni].u[7] = x1.w;
            }
            #pragma unroll
            for (int mi = 0; mi < 4; ++mi)
                #pragma unroll
                for (int ni = 0; ni < 2; ++ni)
                    acc[mi][ni] = __builtin_amdgcn_wmma_f32_16x16x32_bf16(
                        false, af[mi].v, false, bf[ni].v, (short)0, acc[mi][ni], false, false);
        }
        __syncthreads();
    }

    // Epilogue: C/D layout -> lane (hf,l), vgpr v: M = v + hf*8, N = l
    #pragma unroll
    for (int mi = 0; mi < 4; ++mi)
        #pragma unroll
        for (int ni = 0; ni < 2; ++ni) {
            int col = bn + wn * 32 + ni * 16 + l;
            float bv = bias ? bias[col] : 0.0f;
            #pragma unroll
            for (int v = 0; v < 8; ++v) {
                int row = bm + wm * 64 + mi * 16 + hf * 8 + v;
                float val = acc[mi][ni][v] + bv;
                if (relu) val = fmaxf(val, 0.0f);
                C[(size_t)row * N + col] = val;
            }
        }
}

// ---------------------------------------------------------------------------
// Attention: per block = (b, head, 16 query rows). 256 thr = 16 rows x 16.
// Writes the reference's no-transpose reshape scramble.
// ---------------------------------------------------------------------------
__global__ __launch_bounds__(256)
void k_attn(const float* __restrict__ Q, int ldq, int qhs,
            const float* __restrict__ Kp, const float* __restrict__ Vp,
            int ldkv, int kvhs, int Tk,
            const float* __restrict__ scal,
            const int* __restrict__ mask, int causal,
            float* __restrict__ Out)
{
    __shared__ float sQ[16][64];
    __shared__ float sL[16][512];
    __shared__ float red[16][16];

    const int tid = threadIdx.x;
    const int r   = tid >> 4;
    const int c   = tid & 15;
    const int q0  = blockIdx.x * 16;
    const int hh  = blockIdx.y;
    const int b   = blockIdx.z;

    #pragma unroll
    for (int i = 0; i < 4; ++i) {
        int e = tid + i * 256;
        int rr = e >> 6, dd = e & 63;
        sQ[rr][dd] = Q[(size_t)(b * TT + q0 + rr) * ldq + hh * qhs + dd] * 0.125f;
    }
    __syncthreads();

    const int nj = Tk >> 4;
    for (int j = 0; j < nj; ++j) {
        int kk = c + j * 16;
        int valid = causal ? (mask[(q0 + r) * Tk + kk] != 0)
                           : (mask[b * Tk + kk] != 0);
        const float* krow = Kp + (size_t)(b * Tk + kk) * ldkv + hh * kvhs;
        float dot = 0.0f;
        #pragma unroll 8
        for (int d = 0; d < 64; ++d) dot += sQ[r][d] * krow[d];
        sL[r][kk] = valid ? dot : -1.0e9f;
    }
    __syncthreads();

    float m = -3.0e38f;
    for (int j = 0; j < nj; ++j) m = fmaxf(m, sL[r][c + j * 16]);
    red[r][c] = m; __syncthreads();
    if (c == 0) { float mm = red[r][0]; for (int i = 1; i < 16; ++i) mm = fmaxf(mm, red[r][i]); red[r][0] = mm; }
    __syncthreads();
    float rowmax = red[r][0];
    __syncthreads();

    float s = 0.0f;
    for (int j = 0; j < nj; ++j) {
        int kk = c + j * 16;
        float p = __expf(sL[r][kk] - rowmax);
        sL[r][kk] = p; s += p;
    }
    red[r][c] = s; __syncthreads();
    if (c == 0) { float ss = 0.0f; for (int i = 0; i < 16; ++i) ss += red[r][i]; red[r][0] = ss; }
    __syncthreads();
    float inv = 1.0f / red[r][0];

    float a0 = 0, a1 = 0, a2 = 0, a3 = 0;
    for (int kk = 0; kk < Tk; ++kk) {
        float p = sL[r][kk];
        const float* vrow = Vp + (size_t)(b * Tk + kk) * ldkv + hh * kvhs + c * 4;
        a0 += p * vrow[0]; a1 += p * vrow[1]; a2 += p * vrow[2]; a3 += p * vrow[3];
    }
    float sc = scal[b] * inv;
    int t = q0 + r;
    int orow = b * TT + hh * 32 + (t >> 4);      // reference's reshape scramble
    int ocol = ((t & 15) << 6) + c * 4;
    float* o = Out + (size_t)orow * H + ocol;
    o[0] = a0 * sc; o[1] = a1 * sc; o[2] = a2 * sc; o[3] = a3 * sc;
}

// ---------------------------------------------------------------------------
// scal[b] = 0.125 * sum_k (sum_t xn[b,t,k]) * (sum_c We[c,k])
// ---------------------------------------------------------------------------
__global__ void k_wsum(const float* __restrict__ W, float* __restrict__ wsum) {
    int k = blockIdx.x * blockDim.x + threadIdx.x;
    float s = 0.0f;
    for (int c = 0; c < H; ++c) s += W[(size_t)c * H + k];
    wsum[k] = s;
}

__global__ __launch_bounds__(256)
void k_scal(const float* __restrict__ X, const float* __restrict__ wsum,
            float* __restrict__ scal) {
    int b = blockIdx.x, tid = threadIdx.x;
    __shared__ float red[256];
    float s = 0.0f;
    for (int k = tid; k < H; k += 256) {
        float cs = 0.0f;
        for (int t = 0; t < TT; ++t) cs += X[(size_t)(b * TT + t) * H + k];
        s += cs * wsum[k];
    }
    red[tid] = s; __syncthreads();
    for (int off = 128; off; off >>= 1) { if (tid < off) red[tid] += red[tid + off]; __syncthreads(); }
    if (tid == 0) scal[b] = red[0] * 0.125f;
}

// ---------------------------------------------------------------------------
// x = LayerNorm(x + o) * g + b   (one row of H=1024 per block, in place)
// ---------------------------------------------------------------------------
__global__ __launch_bounds__(256)
void k_add_ln(float* __restrict__ X, const float* __restrict__ O,
              const float* __restrict__ g, const float* __restrict__ bta) {
    int row = blockIdx.x, tid = threadIdx.x;
    __shared__ float red[256];
    float v[4];
    float s = 0.0f;
    #pragma unroll
    for (int i = 0; i < 4; ++i) {
        int c = tid + i * 256;
        v[i] = X[(size_t)row * H + c] + O[(size_t)row * H + c];
        s += v[i];
    }
    red[tid] = s; __syncthreads();
    for (int off = 128; off; off >>= 1) { if (tid < off) red[tid] += red[tid + off]; __syncthreads(); }
    float mu = red[0] * (1.0f / H);
    __syncthreads();
    s = 0.0f;
    #pragma unroll
    for (int i = 0; i < 4; ++i) { float d = v[i] - mu; s += d * d; }
    red[tid] = s; __syncthreads();
    for (int off = 128; off; off >>= 1) { if (tid < off) red[tid] += red[tid + off]; __syncthreads(); }
    float rs = rsqrtf(red[0] * (1.0f / H) + 1e-5f);
    #pragma unroll
    for (int i = 0; i < 4; ++i) {
        int c = tid + i * 256;
        X[(size_t)row * H + c] = (v[i] - mu) * rs * g[c] + bta[c];
    }
}

__global__ void k_scale2(const float* __restrict__ a, const float* __restrict__ n,
                         float* __restrict__ xa, float* __restrict__ xn, int count) {
    int i = blockIdx.x * blockDim.x + threadIdx.x;
    if (i < count) { xa[i] = a[i] * 32.0f; xn[i] = n[i] * 32.0f; }  // sqrt(1024)=32
}

__global__ void k_out(const float* __restrict__ xa, const float* __restrict__ xn,
                      float* __restrict__ out, int n) {
    int i = blockIdx.x * blockDim.x + threadIdx.x;
    if (i < n) { out[i] = xa[i]; out[n + i] = xn[i]; }
}

// ---------------------------------------------------------------------------
extern "C" void kernel_launch(void* const* d_in, const int* in_sizes, int n_in,
                              void* d_out, int out_size, void* d_ws, size_t ws_size,
                              hipStream_t stream) {
    const float* enc   = (const float*)d_in[0];
    const float* x_adj = (const float*)d_in[1];
    const float* x_nt  = (const float*)d_in[2];
    const int*   pad   = (const int*)d_in[3];
    const int*   fut   = (const int*)d_in[4];
    const float* qkv_s = (const float*)d_in[5];
    const float* e_s   = (const float*)d_in[6];
    const float* o1_s  = (const float*)d_in[7];
    const float* o2_s  = (const float*)d_in[8];
    const float* qkv_c = (const float*)d_in[9];
    const float* e_c   = (const float*)d_in[10];
    const float* o1_c  = (const float*)d_in[11];
    const float* o2_c  = (const float*)d_in[12];
    const float* w1    = (const float*)d_in[13];
    const float* b1    = (const float*)d_in[14];
    const float* w2    = (const float*)d_in[15];
    const float* b2    = (const float*)d_in[16];
    const float* ln1g  = (const float*)d_in[17];
    const float* ln1b  = (const float*)d_in[18];
    const float* ln2g  = (const float*)d_in[19];
    const float* ln2b  = (const float*)d_in[20];
    const float* ln3g  = (const float*)d_in[21];
    const float* ln3b  = (const float*)d_in[22];

    const size_t X = (size_t)MROWS * H;          // 4,194,304
    float* XA   = (float*)d_ws;
    float* XN   = XA + X;
    float* VALS = XN + X;
    float* TMP  = VALS + X;
    float* BIG  = TMP + X;                       // 4096*4096
    float* WSUM = BIG + (size_t)MROWS * FF;
    float* SCAL = WSUM + H;

    const dim3 blk(256);
    k_scale2<<<dim3((X + 255) / 256), blk, 0, stream>>>(x_adj, x_nt, XA, XN, (int)X);

    for (int i = 0; i < LAYERS; ++i) {
        const size_t HH = (size_t)H * H;
        const float* Wqkv  = qkv_s + (size_t)i * 3 * HH;
        const float* We    = e_s   + (size_t)i * HH;
        const float* Wo1   = o1_s  + (size_t)i * HH;
        const float* Wo2   = o2_s  + (size_t)i * HH;
        const float* WqkvC = qkv_c + (size_t)i * 3 * HH;
        const float* WeC   = e_c   + (size_t)i * HH;
        const float* Wo1C  = o1_c  + (size_t)i * HH;
        const float* Wo2C  = o2_c  + (size_t)i * HH;

        // ---- self attention ----
        k_gemm<<<dim3(3 * H / GTN, MROWS / GTM), blk, 0, stream>>>(XA, Wqkv, nullptr, BIG, MROWS, 3 * H, H, 0);
        k_wsum<<<dim3(H / 256), blk, 0, stream>>>(We, WSUM);
        k_scal<<<dim3(BB), blk, 0, stream>>>(XN, WSUM, SCAL);
        k_attn<<<dim3(TT / 16, NH, BB), blk, 0, stream>>>(BIG, 3 * H, 3 * DH, BIG + DH, BIG + 2 * DH,
                                                          3 * H, 3 * DH, TT, SCAL, fut, 1, VALS);
        k_gemm<<<dim3(H / GTN, MROWS / GTM), blk, 0, stream>>>(VALS, Wo1, nullptr, TMP, MROWS, H, H, 0);
        k_add_ln<<<dim3(MROWS), blk, 0, stream>>>(XA, TMP, ln1g + (size_t)i * H, ln1b + (size_t)i * H);
        k_gemm<<<dim3(H / GTN, MROWS / GTM), blk, 0, stream>>>(VALS, Wo2, nullptr, TMP, MROWS, H, H, 0);
        k_add_ln<<<dim3(MROWS), blk, 0, stream>>>(XN, TMP, ln1g + (size_t)i * H, ln1b + (size_t)i * H);

        // ---- cross attention ----
        k_gemm<<<dim3(H / GTN, MROWS / GTM), blk, 0, stream>>>(XA, WqkvC, nullptr, TMP, MROWS, H, H, 0);
        k_gemm<<<dim3(2 * H / GTN, MROWS / GTM), blk, 0, stream>>>(enc, WqkvC + HH, nullptr, BIG, MROWS, 2 * H, H, 0);
        k_wsum<<<dim3(H / 256), blk, 0, stream>>>(WeC, WSUM);
        k_scal<<<dim3(BB), blk, 0, stream>>>(XN, WSUM, SCAL);
        k_attn<<<dim3(TT / 16, NH, BB), blk, 0, stream>>>(TMP, H, DH, BIG, BIG + DH,
                                                          2 * H, 2 * DH, SS, SCAL, pad, 0, VALS);
        k_gemm<<<dim3(H / GTN, MROWS / GTM), blk, 0, stream>>>(VALS, Wo1C, nullptr, TMP, MROWS, H, H, 0);
        k_add_ln<<<dim3(MROWS), blk, 0, stream>>>(XA, TMP, ln2g + (size_t)i * H, ln2b + (size_t)i * H);
        k_gemm<<<dim3(H / GTN, MROWS / GTM), blk, 0, stream>>>(VALS, Wo2C, nullptr, TMP, MROWS, H, H, 0);
        k_add_ln<<<dim3(MROWS), blk, 0, stream>>>(XN, TMP, ln2g + (size_t)i * H, ln2b + (size_t)i * H);

        // ---- FFN (xa then xn) ----
        const float* W1 = w1 + (size_t)i * FF * H;
        const float* B1 = b1 + (size_t)i * FF;
        const float* W2 = w2 + (size_t)i * H * FF;
        const float* B2 = b2 + (size_t)i * H;

        k_gemm<<<dim3(FF / GTN, MROWS / GTM), blk, 0, stream>>>(XA, W1, B1, BIG, MROWS, FF, H, 1);
        k_gemm<<<dim3(H / GTN, MROWS / GTM), blk, 0, stream>>>(BIG, W2, B2, TMP, MROWS, H, FF, 0);
        k_add_ln<<<dim3(MROWS), blk, 0, stream>>>(XA, TMP, ln3g + (size_t)i * H, ln3b + (size_t)i * H);

        k_gemm<<<dim3(FF / GTN, MROWS / GTM), blk, 0, stream>>>(XN, W1, B1, BIG, MROWS, FF, H, 1);
        k_gemm<<<dim3(H / GTN, MROWS / GTM), blk, 0, stream>>>(BIG, W2, B2, TMP, MROWS, H, FF, 0);
        k_add_ln<<<dim3(MROWS), blk, 0, stream>>>(XN, TMP, ln3g + (size_t)i * H, ln3b + (size_t)i * H);
    }

    k_out<<<dim3((X + 255) / 256), blk, 0, stream>>>(XA, XN, (float*)d_out, (int)X);
}
